// SelfAttention_48473000903563
// MI455X (gfx1250) — compile-verified, matrix operand order checked by reading, not run on hardware
//
#include <hip/hip_runtime.h>
#include <hip/hip_bf16.h>

typedef __attribute__((ext_vector_type(16))) __bf16 v16bf;
typedef __attribute__((ext_vector_type(8)))  float  v8f;

#define NEGV  (-1e12f)

static __device__ inline unsigned short f2bf(float f) {
  unsigned u = __builtin_bit_cast(unsigned, f);
  u += 0x7FFFu + ((u >> 16) & 1u);          // round-to-nearest-even
  return (unsigned short)(u >> 16);
}

// Load a 16x32 (A) or 32x16 (B) bf16 fragment: lane holds row/col (lr), with
// contraction elements {ko..ko+7} and {ko+16..ko+23} -> two 16B loads.
static __device__ inline v16bf load_frag(const unsigned short* p) {
  uint4 a = *(const uint4*)(p);
  uint4 b = *(const uint4*)(p + 16);
  union { uint4 q[2]; v16bf v; } u;
  u.q[0] = a; u.q[1] = b;
  return u.v;
}

static __device__ inline float rmax16(float v) {
  v = fmaxf(v, __shfl_xor(v, 1));
  v = fmaxf(v, __shfl_xor(v, 2));
  v = fmaxf(v, __shfl_xor(v, 4));
  v = fmaxf(v, __shfl_xor(v, 8));
  return v;
}
static __device__ inline float rsum16(float v) {
  v += __shfl_xor(v, 1);
  v += __shfl_xor(v, 2);
  v += __shfl_xor(v, 4);
  v += __shfl_xor(v, 8);
  return v;
}

// ---------------------------------------------------------------- convert
__global__ void cvt_bf16_kernel(const float4* __restrict__ s,
                                uint2* __restrict__ d, int n4) {
  int i = blockIdx.x * blockDim.x + threadIdx.x;
  int stride = gridDim.x * blockDim.x;
  for (; i < n4; i += stride) {
    float4 f = s[i];
    union { unsigned short h[4]; uint2 v; } u;
    u.h[0] = f2bf(f.x); u.h[1] = f2bf(f.y);
    u.h[2] = f2bf(f.z); u.h[3] = f2bf(f.w);
    d[i] = u.v;
  }
}

// ---------------------------------------------------------------- projections
// X[4096 x 1024] bf16 (row-major), W[1024 x 1024] bf16 (row-major [out,in]).
// mode 0: store out[b][h][s][dk]  (Q, K)
// mode 1: store out[b][h][dk][s]  (V transposed)
// Software-pipelined: next k-tile fragments load while current WMMAs issue.
__global__ void proj_kernel(const unsigned short* __restrict__ X,
                            const unsigned short* __restrict__ W,
                            unsigned short* __restrict__ out, int mode) {
  const int S = 2048, H = 16;
  const int m0 = blockIdx.x * 16;
  const int n0 = blockIdx.y * 64;
  const int lane = threadIdx.x & 31;
  const int lr = lane & 15, lg = lane >> 4, ko = lg * 8;

  const unsigned short* Xp = X + (size_t)(m0 + lr) * 1024 + ko;
  const unsigned short* Wp[4];
#pragma unroll
  for (int f = 0; f < 4; ++f) Wp[f] = W + (size_t)(n0 + f * 16 + lr) * 1024 + ko;

  v8f C[4] = {};
  v16bf a0 = load_frag(Xp);
  v16bf b0[4];
#pragma unroll
  for (int f = 0; f < 4; ++f) b0[f] = load_frag(Wp[f]);

#pragma unroll 4
  for (int kk = 0; kk < 1024; kk += 32) {
    const int kn = (kk + 32) & 1023;                 // wraps to 0 on last iter
    v16bf a1 = load_frag(Xp + kn);
    v16bf b1[4];
#pragma unroll
    for (int f = 0; f < 4; ++f) b1[f] = load_frag(Wp[f] + kn);
#pragma unroll
    for (int f = 0; f < 4; ++f)
      C[f] = __builtin_amdgcn_wmma_f32_16x16x32_bf16(false, a0, false, b0[f],
                                                     (short)0, C[f], false, false);
    a0 = a1;
#pragma unroll
    for (int f = 0; f < 4; ++f) b0[f] = b1[f];
  }

#pragma unroll
  for (int f = 0; f < 4; ++f)
#pragma unroll
    for (int i = 0; i < 8; ++i) {
      int gr = m0 + i + 8 * lg;            // b*S + s
      int n  = n0 + f * 16 + lr;           // h*64 + dk
      int b  = gr >> 11, s = gr & 2047;
      int h  = n >> 6,  dk = n & 63;
      size_t dst = (mode == 0)
          ? ((((size_t)b * H + h) * S + s) * 64 + dk)
          : ((((size_t)b * H + h) * 64 + dk) * (size_t)S + s);
      out[dst] = f2bf(C[f][i]);
    }
}

// ---------------------------------------------------------------- attention
// Qb,Kb: [B,H,S,64] bf16 ; Vt: [B,H,64,S] bf16 ; am: [B,S] i32
// one wave per (b,h,16-query tile); flash softmax over 32 keys/iter.
// K frags double-buffered across iterations; V frags issued early so the
// score WMMAs + softmax VALU hide their latency.
__global__ void attn_kernel(const unsigned short* __restrict__ Qb,
                            const unsigned short* __restrict__ Kb,
                            const unsigned short* __restrict__ Vt,
                            const int* __restrict__ am,
                            unsigned short* __restrict__ ctx) {
  const int S = 2048;
  const int bh = blockIdx.x;          // b*16+h
  const int q0 = blockIdx.y * 16;
  const int b  = bh >> 4, h = bh & 15;
  const int lane = threadIdx.x & 31;
  const int lr = lane & 15, lg = lane >> 4, ko = lg * 8;

  __shared__ __align__(16) unsigned short Pl[16 * 32];

  const unsigned short* Kbase = Kb + (size_t)bh * S * 64;
  const unsigned short* Vbase = Vt + (size_t)bh * 64 * S;

  v16bf qa0 = load_frag(Qb + ((size_t)bh * S + q0 + lr) * 64 + ko);
  v16bf qa1 = load_frag(Qb + ((size_t)bh * S + q0 + lr) * 64 + 32 + ko);

  v8f O[4] = {};
  float m_[8], l_[8];
#pragma unroll
  for (int i = 0; i < 8; ++i) { m_[i] = -INFINITY; l_[i] = 0.0f; }

  // preload K fragments for s0 = 0 (frag [t][k]: keys 16t.., dk chunk 32k..)
  v16bf kf[2][2];
#pragma unroll
  for (int t = 0; t < 2; ++t)
#pragma unroll
    for (int k = 0; k < 2; ++k)
      kf[t][k] = load_frag(Kbase + (size_t)(16 * t + lr) * 64 + 32 * k + ko);

  for (int s0 = 0; s0 < S; s0 += 32) {
    // ---- issue V loads for the CURRENT tile (consumed after softmax)
    v16bf vb[4];
#pragma unroll
    for (int f = 0; f < 4; ++f)
      vb[f] = load_frag(Vbase + (size_t)(f * 16 + lr) * S + s0 + ko);

    const int key0 = s0 + lr, key1 = s0 + 16 + lr;
    const int a0 = am[b * S + key0];
    const int a1 = am[b * S + key1];

    // ---- prefetch two tiles ahead
    {
      const int sp = (s0 + 64) & (S - 1);
      __builtin_prefetch(Kbase + (size_t)(sp + lr) * 64);
      __builtin_prefetch(Vbase + (size_t)lr * S + sp);
    }

    // ---- scores: Q(16x64) . K^T(64x32) -> two 16x16 f32 frags
    v8f sc[2];
#pragma unroll
    for (int t = 0; t < 2; ++t) {
      v8f c = {};
      c = __builtin_amdgcn_wmma_f32_16x16x32_bf16(false, qa0, false, kf[t][0],
                                                  (short)0, c, false, false);
      c = __builtin_amdgcn_wmma_f32_16x16x32_bf16(false, qa1, false, kf[t][1],
                                                  (short)0, c, false, false);
      sc[t] = c;
    }

    // ---- issue K loads for the NEXT tile (overlap with softmax VALU)
    const int sn = (s0 + 32) & (S - 1);
    v16bf kn[2][2];
#pragma unroll
    for (int t = 0; t < 2; ++t)
#pragma unroll
      for (int k = 0; k < 2; ++k)
        kn[t][k] = load_frag(Kbase + (size_t)(sn + 16 * t + lr) * 64 + 32 * k + ko);

    // ---- scale + faithful mask: allowed iff am[key] && key > q
#pragma unroll
    for (int i = 0; i < 8; ++i) {
      int q = q0 + i + 8 * lg;
      float v0 = sc[0][i] * 0.125f;
      float v1 = sc[1][i] * 0.125f;
      if (!(a0 && key0 > q)) v0 += NEGV;
      if (!(a1 && key1 > q)) v1 += NEGV;
      sc[0][i] = v0; sc[1][i] = v1;
    }
    // ---- online softmax (row stats per C-frag row, 16-lane reductions)
#pragma unroll
    for (int i = 0; i < 8; ++i) {
      float rm = rmax16(fmaxf(sc[0][i], sc[1][i]));
      float mn = fmaxf(m_[i], rm);
      float corr = __expf(m_[i] - mn);
      float p0 = __expf(sc[0][i] - mn);
      float p1 = __expf(sc[1][i] - mn);
      float rs = rsum16(p0 + p1);
      l_[i] = l_[i] * corr + rs;
      m_[i] = mn;
#pragma unroll
      for (int f = 0; f < 4; ++f) O[f][i] *= corr;
      int row = i + 8 * lg;
      Pl[row * 32 + lr]      = f2bf(p0);
      Pl[row * 32 + 16 + lr] = f2bf(p1);
    }
    asm volatile("s_wait_dscnt 0" ::: "memory");  // in-wave C->A transpose via LDS
    // ---- PV: P(16x32) . V(32x64)
    v16bf pa = load_frag(&Pl[lr * 32 + ko]);
#pragma unroll
    for (int f = 0; f < 4; ++f)
      O[f] = __builtin_amdgcn_wmma_f32_16x16x32_bf16(false, pa, false, vb[f],
                                                     (short)0, O[f], false, false);
    // rotate K buffers
#pragma unroll
    for (int t = 0; t < 2; ++t)
#pragma unroll
      for (int k = 0; k < 2; ++k)
        kf[t][k] = kn[t][k];
  }
  // ---- normalize + store ctx [B,S,H*64] bf16
#pragma unroll
  for (int i = 0; i < 8; ++i) {
    float inv = 1.0f / l_[i];
    int q = q0 + i + 8 * lg;
#pragma unroll
    for (int f = 0; f < 4; ++f) {
      size_t dst = ((size_t)(b * S + q)) * 1024 + h * 64 + f * 16 + lr;
      ctx[dst] = f2bf(O[f][i] * inv);
    }
  }
}

// ---------------------------------------------------------------- out proj + residual + LN
// ctx [4096x1024] bf16, Wo [1024x1024] bf16; 16 waves/block (512 thr),
// 16 rows/block, 64 cols/wave. Software-pipelined like proj_kernel.
__global__ void out_ln_kernel(const unsigned short* __restrict__ ctx,
                              const unsigned short* __restrict__ Wo,
                              const float* __restrict__ resid,
                              const float* __restrict__ gamma,
                              const float* __restrict__ beta,
                              float* __restrict__ out) {
  const int m0 = blockIdx.x * 16;
  const int w = threadIdx.x >> 5;
  const int lane = threadIdx.x & 31;
  const int lr = lane & 15, lg = lane >> 4, ko = lg * 8;
  const int n0 = w * 64;

  __shared__ float psum[16][16];
  __shared__ float psq[16][16];

  const unsigned short* Ap = ctx + (size_t)(m0 + lr) * 1024 + ko;
  const unsigned short* Bp[4];
#pragma unroll
  for (int f = 0; f < 4; ++f) Bp[f] = Wo + (size_t)(n0 + f * 16 + lr) * 1024 + ko;

  v8f C[4] = {};
  v16bf a0 = load_frag(Ap);
  v16bf b0[4];
#pragma unroll
  for (int f = 0; f < 4; ++f) b0[f] = load_frag(Bp[f]);

#pragma unroll 4
  for (int kk = 0; kk < 1024; kk += 32) {
    const int knx = (kk + 32) & 1023;
    v16bf a1 = load_frag(Ap + knx);
    v16bf b1[4];
#pragma unroll
    for (int f = 0; f < 4; ++f) b1[f] = load_frag(Bp[f] + knx);
#pragma unroll
    for (int f = 0; f < 4; ++f)
      C[f] = __builtin_amdgcn_wmma_f32_16x16x32_bf16(false, a0, false, b0[f],
                                                     (short)0, C[f], false, false);
    a0 = a1;
#pragma unroll
    for (int f = 0; f < 4; ++f) b0[f] = b1[f];
  }

  // residual
#pragma unroll
  for (int f = 0; f < 4; ++f)
#pragma unroll
    for (int i = 0; i < 8; ++i) {
      int row = m0 + i + 8 * lg;
      int col = n0 + f * 16 + lr;
      C[f][i] += resid[(size_t)row * 1024 + col];
    }
  // per-wave partial row stats
#pragma unroll
  for (int i = 0; i < 8; ++i) {
    float ps = 0.f, pq = 0.f;
#pragma unroll
    for (int f = 0; f < 4; ++f) { float v = C[f][i]; ps += v; pq += v * v; }
    ps = rsum16(ps); pq = rsum16(pq);
    if (lr == 0) { psum[w][i + 8 * lg] = ps; psq[w][i + 8 * lg] = pq; }
  }
  __syncthreads();
  float mean[8], inv[8];
#pragma unroll
  for (int i = 0; i < 8; ++i) {
    int r = i + 8 * lg;
    float s = 0.f, q = 0.f;
#pragma unroll
    for (int ww = 0; ww < 16; ++ww) { s += psum[ww][r]; q += psq[ww][r]; }
    float mu = s * (1.0f / 1024.0f);
    float var = q * (1.0f / 1024.0f) - mu * mu;
    mean[i] = mu;
    inv[i] = rsqrtf(var + 1e-5f);
  }
#pragma unroll
  for (int f = 0; f < 4; ++f)
#pragma unroll
    for (int i = 0; i < 8; ++i) {
      int row = m0 + i + 8 * lg;
      int col = n0 + f * 16 + lr;
      out[(size_t)row * 1024 + col] =
          (C[f][i] - mean[i]) * inv[i] * gamma[col] + beta[col];
    }
}

// ---------------------------------------------------------------- launch
extern "C" void kernel_launch(void* const* d_in, const int* in_sizes, int n_in,
                              void* d_out, int out_size, void* d_ws, size_t ws_size,
                              hipStream_t stream) {
  (void)in_sizes; (void)n_in; (void)out_size; (void)ws_size;
  const float* Qf = (const float*)d_in[0];
  const float* Kf = (const float*)d_in[1];
  const float* Vf = (const float*)d_in[2];
  const int*   am = (const int*)  d_in[3];
  const float* Wq = (const float*)d_in[4];
  const float* Wk = (const float*)d_in[5];
  const float* Wv = (const float*)d_in[6];
  const float* Wo = (const float*)d_in[7];
  const float* ga = (const float*)d_in[8];
  const float* be = (const float*)d_in[9];
  float* out = (float*)d_out;

  const size_t NX = 4096ull * 1024;   // B*S*D elems
  const size_t NW = 1024ull * 1024;   // weight elems
  unsigned short* ws = (unsigned short*)d_ws;
  unsigned short* Xq  = ws;            // bf16 inputs
  unsigned short* Xk  = Xq  + NX;
  unsigned short* Xv  = Xk  + NX;
  unsigned short* Wqb = Xv  + NX;      // bf16 weights
  unsigned short* Wkb = Wqb + NW;
  unsigned short* Wvb = Wkb + NW;
  unsigned short* Wob = Wvb + NW;
  unsigned short* Qb  = Wob + NW;      // [B,H,S,64]
  unsigned short* Kb  = Qb  + NX;      // [B,H,S,64]
  unsigned short* Vt  = Kb  + NX;      // [B,H,64,S]
  unsigned short* Ctx = Vt  + NX;      // [B,S,1024]

  cvt_bf16_kernel<<<1024, 256, 0, stream>>>((const float4*)Qf, (uint2*)Xq,  (int)(NX >> 2));
  cvt_bf16_kernel<<<1024, 256, 0, stream>>>((const float4*)Kf, (uint2*)Xk,  (int)(NX >> 2));
  cvt_bf16_kernel<<<1024, 256, 0, stream>>>((const float4*)Vf, (uint2*)Xv,  (int)(NX >> 2));
  cvt_bf16_kernel<<< 256, 256, 0, stream>>>((const float4*)Wq, (uint2*)Wqb, (int)(NW >> 2));
  cvt_bf16_kernel<<< 256, 256, 0, stream>>>((const float4*)Wk, (uint2*)Wkb, (int)(NW >> 2));
  cvt_bf16_kernel<<< 256, 256, 0, stream>>>((const float4*)Wv, (uint2*)Wvb, (int)(NW >> 2));
  cvt_bf16_kernel<<< 256, 256, 0, stream>>>((const float4*)Wo, (uint2*)Wob, (int)(NW >> 2));

  dim3 gp(256, 16);
  proj_kernel<<<gp, 32, 0, stream>>>(Xq, Wqb, Qb, 0);
  proj_kernel<<<gp, 32, 0, stream>>>(Xk, Wkb, Kb, 0);
  proj_kernel<<<gp, 32, 0, stream>>>(Xv, Wvb, Vt, 1);

  dim3 gattn(32, 128);
  attn_kernel<<<gattn, 32, 0, stream>>>(Qb, Kb, Vt, am, Ctx);

  out_ln_kernel<<<256, 512, 0, stream>>>(Ctx, Wob, Qf, ga, be, out);
}
